// NoisyTopK_40295383171124
// MI455X (gfx1250) — matrix operand order repaired and checked by reading, not compiled, not run on hardware
//
#include <hip/hip_runtime.h>
#include <hip/hip_bf16.h>
#include <math.h>
#include <stdint.h>

// ---------------------------------------------------------------------------
// NoisyTopK router for MI455X (gfx1250, wave32).
//   noisy = x @ (W1+W2)^T + (b1+b2);  top8 -> masked softmax.
// GEMM: M=16384 tokens, K=2048, N=64 experts via v_wmma_f32_16x16x32_bf16.
// B double-buffered in LDS via global_load_async_to_lds_b128 (ASYNCcnt).
// ---------------------------------------------------------------------------

typedef __attribute__((ext_vector_type(16))) __bf16 v16bf;
typedef __attribute__((ext_vector_type(2)))  __bf16 v2bf;
typedef __attribute__((ext_vector_type(8)))  float  v8f;
typedef int v4i_g __attribute__((vector_size(4 * sizeof(int))));  // builtin's param type

#define TOKENS   16384
#define DIM      2048
#define NEXP     64
#define TOPK     8
#define BLOCK    128      // 4 waves
#define WAVES    4
#define KCHUNK   256      // K per LDS stage (8 WMMA K-steps)
#define NCHUNKS  (DIM / KCHUNK)

#if defined(__HIP_DEVICE_COMPILE__) &&                                   \
    __has_builtin(__builtin_amdgcn_global_load_async_to_lds_b128) &&     \
    __has_builtin(__builtin_amdgcn_s_wait_asynccnt)
#define USE_ASYNC_LDS 1
#else
#define USE_ASYNC_LDS 0
#endif

__device__ __forceinline__ unsigned short f2bf_bits(float f) {
  union { float f; unsigned u; } v; v.f = f;
  unsigned r = v.u + 0x7FFFu + ((v.u >> 16) & 1u);   // round-to-nearest-even
  return (unsigned short)(r >> 16);
}

__device__ __forceinline__ v2bf pk_bf16(float x, float y) {
#if defined(__HIP_DEVICE_COMPILE__) && __has_builtin(__builtin_amdgcn_cvt_pk_bf16_f32)
  return __builtin_amdgcn_cvt_pk_bf16_f32(x, y);     // v_cvt_pk_bf16_f32
#else
  v2bf r; r[0] = (__bf16)x; r[1] = (__bf16)y; return r;
#endif
}

// ---------------------------------------------------------------------------
// Prep: Wc = W1+W2 in bf16, pre-swizzled into WMMA B-fragment order.
// Layout: frag[(((chunk*4 + ntile)*8 + s_local)*32 + lane)*16 + e]
//   where k = (chunk*8+s_local)*32 + (e<8 ? (lane>>4)*8+e : 16+(lane>>4)*8+e-8), col = ntile*16 + (lane&15)
// (16-bit B 32x16: lanes 0-15 hold K=0..7 & 16..23, lanes 16-31 K=8..15 & 24..31,
//  matching the A fragment order; column = lane&15).
// ---------------------------------------------------------------------------
__global__ __launch_bounds__(256) void prep_kernel(
    const float* __restrict__ W1, const float* __restrict__ b1,
    const float* __restrict__ W2, const float* __restrict__ b2,
    unsigned short* __restrict__ frag, float* __restrict__ bc) {
  int t = blockIdx.x * 256 + threadIdx.x;
  if (t < NEXP) bc[t] = b1[t] + b2[t];
  if (t >= NEXP * DIM) return;
  int e    = t & 15;
  int lane = (t >> 4) & 31;
  int rest = t >> 9;
  int sl   = rest & 7;  rest >>= 3;
  int nt   = rest & 3;
  int ch   = rest >> 2;
  int s    = ch * 8 + sl;
  int k    = s * 32 + (e < 8 ? (lane >> 4) * 8 + e : 16 + (lane >> 4) * 8 + (e - 8));
  int col  = nt * 16 + (lane & 15);
  float w  = W1[col * DIM + k] + W2[col * DIM + k];
  frag[t]  = f2bf_bits(w);
}

// ---------------------------------------------------------------------------
// Main: 4 waves/block, each wave = one 16-token tile x all 64 experts.
// ---------------------------------------------------------------------------
__global__ __launch_bounds__(BLOCK) void router_kernel(
    const float* __restrict__ x,
    const unsigned short* __restrict__ frag,
    const float* __restrict__ bc,
    float* __restrict__ probs,
    int*   __restrict__ topk_idx) {
  __shared__ uint4 lds4[4096];                 // 64 KB: 2 x 32 KB B-chunk buffers
  const int tid  = threadIdx.x;
  const int wave = tid >> 5;
  const int lane = tid & 31;
  const int half = lane >> 4;                  // 0: lanes 0-15, 1: lanes 16-31
  const int m    = lane & 15;
  const size_t row0 = ((size_t)blockIdx.x * WAVES + wave) * 16;
  const float* xrow = x + (row0 + m) * DIM;

  // stage chunk c of pre-swizzled B into LDS buffer (c&1), async if available
  auto stage = [&](int c) {
    const uint4* src = (const uint4*)frag + c * 2048;
    uint4* dst = lds4 + (c & 1) * 2048;
#if USE_ASYNC_LDS
    #pragma unroll
    for (int i = 0; i < 16; ++i) {
      const uint4* s = src + tid + i * BLOCK;
      uint4* d = dst + tid + i * BLOCK;
      __builtin_amdgcn_global_load_async_to_lds_b128(
          (__attribute__((address_space(1))) v4i_g*)(uintptr_t)s,
          (__attribute__((address_space(3))) v4i_g*)(unsigned)(uintptr_t)d,
          0, 0);
    }
#else
    #pragma unroll
    for (int i = 0; i < 16; ++i)
      dst[tid + i * BLOCK] = src[tid + i * BLOCK];
#endif
  };

  v8f acc0 = {}, acc1 = {}, acc2 = {}, acc3 = {};

  stage(0);                                    // prologue prefetch

  for (int c = 0; c < NCHUNKS; ++c) {
#if USE_ASYNC_LDS
    __builtin_amdgcn_s_wait_asynccnt(0);       // my slice of buf[c&1] landed
#endif
    __syncthreads();                           // whole buf[c&1] ready; buf[!c&1] free
    if (c + 1 < NCHUNKS) stage(c + 1);         // overlap next fetch with compute

    const char* lb = (const char*)(lds4 + (c & 1) * 2048);
    #pragma unroll
    for (int sl = 0; sl < 8; ++sl) {
      const int k0 = c * KCHUNK + sl * 32;
      // A fragment: 16-bit A 16x32 layout. half=0 -> K 0-7 & 16-23,
      // half=1 -> K 8-15 & 24-31 (token = lane&15).
      const int ka = k0 + half * 8;
      const int kb = k0 + 16 + half * 8;
      float4 a0 = *(const float4*)(xrow + ka);
      float4 a1 = *(const float4*)(xrow + ka + 4);
      float4 a2 = *(const float4*)(xrow + kb);
      float4 a3 = *(const float4*)(xrow + kb + 4);
      union { v2bf p[8]; v16bf v; } A;
      A.p[0] = pk_bf16(a0.x, a0.y);  A.p[1] = pk_bf16(a0.z, a0.w);
      A.p[2] = pk_bf16(a1.x, a1.y);  A.p[3] = pk_bf16(a1.z, a1.w);
      A.p[4] = pk_bf16(a2.x, a2.y);  A.p[5] = pk_bf16(a2.z, a2.w);
      A.p[6] = pk_bf16(a3.x, a3.y);  A.p[7] = pk_bf16(a3.z, a3.w);

      v16bf b0 = *(const v16bf*)(lb + (((0*8 + sl)*32 + lane) << 5));
      v16bf b1 = *(const v16bf*)(lb + (((1*8 + sl)*32 + lane) << 5));
      v16bf b2 = *(const v16bf*)(lb + (((2*8 + sl)*32 + lane) << 5));
      v16bf b3 = *(const v16bf*)(lb + (((3*8 + sl)*32 + lane) << 5));

      acc0 = __builtin_amdgcn_wmma_f32_16x16x32_bf16(false, A.v, false, b0,
                                                     (short)0, acc0, false, false);
      acc1 = __builtin_amdgcn_wmma_f32_16x16x32_bf16(false, A.v, false, b1,
                                                     (short)0, acc1, false, false);
      acc2 = __builtin_amdgcn_wmma_f32_16x16x32_bf16(false, A.v, false, b2,
                                                     (short)0, acc2, false, false);
      acc3 = __builtin_amdgcn_wmma_f32_16x16x32_bf16(false, A.v, false, b3,
                                                     (short)0, acc3, false, false);
    }
  }

  // --- transpose accumulators (+bias) into per-token logit rows in LDS ---
  // C/D layout: lanes 0-15: N=lane, VGPR r -> M=r; lanes 16-31: N=lane-16, M=8+r.
  // Buffer 0 is safe to reuse: its last readers finished before the final
  // iteration's top barrier; regions below are per-wave disjoint.
  float* wlog = (float*)lds4 + wave * (16 * NEXP);   // 4 waves * 4KB = 16KB
  #pragma unroll
  for (int nt = 0; nt < 4; ++nt) {
    const int eidx = nt * 16 + m;
    const float b  = bc[eidx];
    const v8f acc  = (nt == 0) ? acc0 : (nt == 1) ? acc1 : (nt == 2) ? acc2 : acc3;
    #pragma unroll
    for (int r = 0; r < 8; ++r) {
      const int tok = half * 8 + r;
      wlog[tok * NEXP + eidx] = acc[r] + b;
    }
  }
  __syncthreads();

  // --- top-8 + masked softmax: lanes 0-15 each own one token ---
  if (lane < 16) {
    const int tok = lane;
    float rowv[NEXP];
    #pragma unroll
    for (int e = 0; e < NEXP; ++e) rowv[e] = wlog[tok * NEXP + e];

    float tv[TOPK]; int ti[TOPK];
    float pv = INFINITY; int pi = -1;          // (value desc, index asc) order
    for (int j = 0; j < TOPK; ++j) {
      float mval = -INFINITY; int mi = 0;
      #pragma unroll
      for (int e = 0; e < NEXP; ++e) {
        const float v = rowv[e];
        const bool after = (v < pv) || (v == pv && e > pi);
        if (after && v > mval) { mval = v; mi = e; }
      }
      tv[j] = mval; ti[j] = mi; pv = mval; pi = mi;
    }

    // softmax over the 8 kept logits (rest are -inf -> prob 0)
    float s = 0.f, p[TOPK];
    #pragma unroll
    for (int j = 0; j < TOPK; ++j) { p[j] = __expf(tv[j] - tv[0]); s += p[j]; }
    const float inv = 1.f / s;

    // build sparse prob row in LDS, then vector-store
    #pragma unroll
    for (int e = 0; e < NEXP; ++e) wlog[tok * NEXP + e] = 0.f;
    #pragma unroll
    for (int j = 0; j < TOPK; ++j) wlog[tok * NEXP + ti[j]] = p[j] * inv;

    const size_t g = row0 + tok;
    float4* orow = (float4*)(probs + g * NEXP);
    const float4* lrow = (const float4*)(wlog + tok * NEXP);
    #pragma unroll
    for (int q = 0; q < NEXP / 4; ++q) orow[q] = lrow[q];
    #pragma unroll
    for (int j = 0; j < TOPK; ++j) topk_idx[g * TOPK + j] = ti[j];
  }
}

// ---------------------------------------------------------------------------
extern "C" void kernel_launch(void* const* d_in, const int* in_sizes, int n_in,
                              void* d_out, int out_size, void* d_ws, size_t ws_size,
                              hipStream_t stream) {
  const float* x  = (const float*)d_in[0];
  const float* W1 = (const float*)d_in[1];
  const float* b1 = (const float*)d_in[2];
  const float* W2 = (const float*)d_in[3];
  const float* b2 = (const float*)d_in[4];

  unsigned short* frag = (unsigned short*)d_ws;                 // 256 KB
  float* bc = (float*)((char*)d_ws + (size_t)NEXP * DIM * 2);   // 256 B

  float* probs = (float*)d_out;                                 // [16384, 64]
  int*   idx   = (int*)(probs + (size_t)TOKENS * NEXP);         // [16384, 8]

  prep_kernel<<<(NEXP * DIM + 255) / 256, 256, 0, stream>>>(W1, b1, W2, b2, frag, bc);
  router_kernel<<<TOKENS / (16 * WAVES), BLOCK, 0, stream>>>(x, frag, bc, probs, idx);
}
